// AttentionLayer_7086696038681
// MI455X (gfx1250) — matrix operand order repaired and checked by reading, not compiled, not run on hardware
//
#include <hip/hip_runtime.h>
#include <hip/hip_bf16.h>

typedef __bf16 bf16;
typedef __attribute__((ext_vector_type(16))) __bf16 v16bf;
typedef __attribute__((ext_vector_type(8)))  float  v8f;

typedef int vint4 __attribute__((ext_vector_type(4)));
typedef __attribute__((address_space(1))) vint4 g_vint4;  // HIP "__device__" AS
typedef __attribute__((address_space(3))) vint4 l_vint4;  // HIP "__shared__" AS

union AFrag { v16bf v; unsigned u[8]; };

// ---------------- async global->LDS copy (gfx1250 ASYNCcnt path) -----------
#if __has_builtin(__builtin_amdgcn_global_load_async_to_lds_b128)
#define HAVE_ASYNC_LDS 1
__device__ inline void async_cp_b128(const void* g, void* l) {
  __builtin_amdgcn_global_load_async_to_lds_b128(
      (g_vint4*)g, (l_vint4*)l, 0, 0);
}
#elif __has_builtin(__builtin_amdgcn_global_async_load_to_lds_b128)
#define HAVE_ASYNC_LDS 1
__device__ inline void async_cp_b128(const void* g, void* l) {
  __builtin_amdgcn_global_async_load_to_lds_b128(
      (g_vint4*)g, (l_vint4*)l, 0, 0);
}
#else
#define HAVE_ASYNC_LDS 0
__device__ inline void async_cp_b128(const void* g, void* l) {
  *reinterpret_cast<uint4*>(l) = *reinterpret_cast<const uint4*>(g);
}
#endif

__device__ inline void wait_async_le4() {
#if __has_builtin(__builtin_amdgcn_s_wait_asynccnt)
  __builtin_amdgcn_s_wait_asynccnt(4);
#else
  asm volatile("s_wait_asynccnt 4" ::: "memory");
#endif
}
__device__ inline void wait_async_le0() {
#if __has_builtin(__builtin_amdgcn_s_wait_asynccnt)
  __builtin_amdgcn_s_wait_asynccnt(0);
#else
  asm volatile("s_wait_asynccnt 0" ::: "memory");
#endif
}

// A-matrix 16x32 bf16 fragment from row-major storage (stride in bf16 pairs).
__device__ inline AFrag load_a16(const unsigned* p, int row_stride_pairs,
                                 int row0, int col_pair_off) {
  const int lane = threadIdx.x & 31;
  const int m = lane & 15;
  const int h = (lane >> 4) << 2;  // 0 or 4 pairs
  const unsigned* r = p + (size_t)(row0 + m) * row_stride_pairs + col_pair_off + h;
  AFrag a;
#pragma unroll
  for (int j = 0; j < 4; ++j) a.u[j] = r[j];
#pragma unroll
  for (int j = 0; j < 4; ++j) a.u[4 + j] = r[8 + j];
  return a;
}

// B-matrix 32x16 bf16 fragment; storage is [n][k] rows of K values.
__device__ inline AFrag load_b16(const unsigned* p, int col_stride_pairs,
                                 int n0, int k_pair_off) {
  const int lane = threadIdx.x & 31;
  const int n = n0 + (lane & 15);
  const int h = (lane >> 4) << 2;  // 0 or 4 pairs
  const unsigned* r = p + (size_t)n * col_stride_pairs + k_pair_off + h;
  AFrag b;
#pragma unroll
  for (int j = 0; j < 4; ++j) b.u[j] = r[j];
#pragma unroll
  for (int j = 0; j < 4; ++j) b.u[4 + j] = r[8 + j];
  return b;
}

__device__ inline v8f wmma_bf16(const AFrag& a, const AFrag& b, v8f c) {
  return __builtin_amdgcn_wmma_f32_16x16x32_bf16(
      false, a.v, false, b.v, (short)0, c, false, false);
}

// ---------------------------------------------------------------------------
// Projection: O[row, e] = sum_d X[row, d] * W[e, d]  (x @ W^T), bf16 out.
// For q (which==0) the softmax scale 1/sqrt(D) and log2(e) are folded in so
// the attention hot loop can use raw exp2.
// ---------------------------------------------------------------------------
__global__ __launch_bounds__(256) void proj_kernel(
    const float* __restrict__ Xq, const float* __restrict__ Xk, const float* __restrict__ Xv,
    const float* __restrict__ Wq, const float* __restrict__ Wk, const float* __restrict__ Wv,
    bf16* __restrict__ Oq, bf16* __restrict__ Ok, bf16* __restrict__ Ov) {
  const int which = blockIdx.y;
  const float* X = (which == 0) ? Xq : (which == 1) ? Xk : Xv;
  const float* W = (which == 0) ? Wq : (which == 1) ? Wk : Wv;
  bf16* O = (which == 0) ? Oq : (which == 1) ? Ok : Ov;
  const float oscale = (which == 0) ? 0.09016844005556021f : 1.0f;  // (1/16)*log2(e)

  __shared__ bf16 Xs[128 * 32];
  __shared__ bf16 Ws[256 * 32];

  const int tid = threadIdx.x;
  const int wave = tid >> 5;
  const int lane = tid & 31;
  const size_t row0 = (size_t)blockIdx.x * 128;

  v8f acc[16];
#pragma unroll
  for (int n = 0; n < 16; ++n) {
    v8f z = {0.f, 0.f, 0.f, 0.f, 0.f, 0.f, 0.f, 0.f};
    acc[n] = z;
  }

#pragma unroll 1
  for (int c = 0; c < 8; ++c) {
#pragma unroll
    for (int k = 0; k < 4; ++k) {
      int p4 = tid + k * 256;
      int row = p4 >> 3;
      int col = (p4 & 7) << 2;
      const float4 f =
          *reinterpret_cast<const float4*>(X + (row0 + row) * 256 + c * 32 + col);
      bf16* d = &Xs[row * 32 + col];
      d[0] = (bf16)f.x; d[1] = (bf16)f.y; d[2] = (bf16)f.z; d[3] = (bf16)f.w;
    }
#pragma unroll
    for (int k = 0; k < 8; ++k) {
      int p4 = tid + k * 256;
      int e = p4 >> 3;
      int col = (p4 & 7) << 2;
      const float4 f =
          *reinterpret_cast<const float4*>(W + (size_t)e * 256 + c * 32 + col);
      bf16* d = &Ws[e * 32 + col];
      d[0] = (bf16)f.x; d[1] = (bf16)f.y; d[2] = (bf16)f.z; d[3] = (bf16)f.w;
    }
    __syncthreads();

    AFrag a = load_a16(reinterpret_cast<const unsigned*>(Xs), 16, wave * 16, 0);
#pragma unroll
    for (int n = 0; n < 16; ++n) {
      AFrag b = load_b16(reinterpret_cast<const unsigned*>(Ws), 16, n * 16, 0);
      acc[n] = wmma_bf16(a, b, acc[n]);
    }
    __syncthreads();
  }

  const int rowl_base = (lane >> 4) << 3;
#pragma unroll
  for (int n = 0; n < 16; ++n) {
#pragma unroll
    for (int r = 0; r < 8; ++r) {
      int rowl = r + rowl_base;
      int col = n * 16 + (lane & 15);
      O[(row0 + wave * 16 + rowl) * 256 + col] = (bf16)(acc[n][r] * oscale);
    }
  }
}

// ---------------------------------------------------------------------------
// Flash attention, double-buffered K via async global->LDS, V software-
// pipelined through registers (transposed into LDS), online softmax in f32.
// Grid: B * S/128 blocks of 256 threads; each wave owns 16 query rows.
// ---------------------------------------------------------------------------
__global__ __launch_bounds__(256) void attn_kernel(
    const bf16* __restrict__ Q, const bf16* __restrict__ K,
    const bf16* __restrict__ V, float* __restrict__ Out) {
  constexpr int S = 4096, T = 4096, D = 256, NT = T / 32;

  __shared__ bf16 Ks[2][32 * 256];  // [key][d], double buffered (async dest)
  __shared__ bf16 Vs[256 * 32];     // [d][key] transposed, reg-pipelined
  __shared__ bf16 Ps[8][16 * 32];   // per-wave P staging [row][key]

  const int tid = threadIdx.x;
  const int wave = tid >> 5;
  const int lane = tid & 31;
  const int b = blockIdx.x >> 5;
  const int qblk = blockIdx.x & 31;
  const bf16* Qb = Q + (size_t)b * S * D;
  const bf16* Kb = K + (size_t)b * T * D;
  const bf16* Vb = V + (size_t)b * T * D;
  const int rbase = qblk * 128 + wave * 16;
  const int rowl_base = (lane >> 4) << 3;

  AFrag qa[8];
#pragma unroll
  for (int c = 0; c < 8; ++c)
    qa[c] = load_a16(reinterpret_cast<const unsigned*>(Qb), 128, rbase, c * 16);

  v8f o[16];
#pragma unroll
  for (int n = 0; n < 16; ++n) {
    v8f z = {0.f, 0.f, 0.f, 0.f, 0.f, 0.f, 0.f, 0.f};
    o[n] = z;
  }
  float mrow[8], lrow[8];
#pragma unroll
  for (int r = 0; r < 8; ++r) { mrow[r] = -3.0e38f; lrow[r] = 0.f; }

  const unsigned* Vg = reinterpret_cast<const unsigned*>(Vb);
  const char* Kg8 = reinterpret_cast<const char*>(Kb);

  // ---- preamble: tile 0 in flight ----
  unsigned vreg[16];
#pragma unroll
  for (int i = 0; i < 4; ++i) {
    int g = tid + i * 256;  // 16B granules of the 16KB K tile
    async_cp_b128(Kg8 + (size_t)g * 16, ((char*)&Ks[0][0]) + g * 16);
  }
#pragma unroll
  for (int k = 0; k < 16; ++k) vreg[k] = Vg[tid + k * 256];

#pragma unroll 1
  for (int kt = 0; kt < NT; ++kt) {
    const int buf = kt & 1;

    // 1. Commit this tile's V registers into LDS, transposed to [d][key].
#pragma unroll
    for (int k = 0; k < 16; ++k) {
      int p = tid + k * 256;
      union { unsigned u; bf16 h[2]; } cv;
      cv.u = vreg[k];
      int key = p >> 7;
      int d0 = (p & 127) << 1;
      Vs[d0 * 32 + key] = cv.h[0];
      Vs[(d0 + 1) * 32 + key] = cv.h[1];
    }

    // 2. Issue next tile: async K copies + V global loads into registers.
    if (kt + 1 < NT) {
      const size_t nbyte = (size_t)(kt + 1) * 32 * 256 * 2;
#pragma unroll
      for (int i = 0; i < 4; ++i) {
        int g = tid + i * 256;
        async_cp_b128(Kg8 + nbyte + (size_t)g * 16,
                      ((char*)&Ks[1 - buf][0]) + g * 16);
      }
      const size_t nelem = (size_t)(kt + 1) * 32 * 128;  // uint granules
#pragma unroll
      for (int k = 0; k < 16; ++k) vreg[k] = Vg[nelem + tid + k * 256];
      wait_async_le4();  // current tile's 4 asyncs are older -> complete
    } else {
      wait_async_le0();
    }
    __syncthreads();  // Ks[buf] + Vs now valid for every wave

    // 3. S = Q @ K^T (16 rows x 32 keys), f32 accumulate.
    v8f s0 = {0.f, 0.f, 0.f, 0.f, 0.f, 0.f, 0.f, 0.f};
    v8f s1 = {0.f, 0.f, 0.f, 0.f, 0.f, 0.f, 0.f, 0.f};
    const unsigned* Ku = reinterpret_cast<const unsigned*>(&Ks[buf][0]);
#pragma unroll
    for (int c = 0; c < 8; ++c) {
      AFrag b0 = load_b16(Ku, 128, 0, c * 16);
      s0 = wmma_bf16(qa[c], b0, s0);
      AFrag b1 = load_b16(Ku, 128, 16, c * 16);
      s1 = wmma_bf16(qa[c], b1, s1);
    }

    // 4. Online softmax (base-2; scale folded into q at projection time).
    float tmax[8];
#pragma unroll
    for (int r = 0; r < 8; ++r) tmax[r] = fmaxf(s0[r], s1[r]);
#pragma unroll
    for (int ms = 8; ms >= 1; ms >>= 1)
#pragma unroll
      for (int r = 0; r < 8; ++r)
        tmax[r] = fmaxf(tmax[r], __shfl_xor(tmax[r], ms, 32));

    float tsum[8];
#pragma unroll
    for (int r = 0; r < 8; ++r) {
      float mn = fmaxf(mrow[r], tmax[r]);
      float alpha = exp2f(mrow[r] - mn);
      mrow[r] = mn;
      float p0 = exp2f(s0[r] - mn);
      float p1 = exp2f(s1[r] - mn);
      tsum[r] = p0 + p1;
      lrow[r] *= alpha;
#pragma unroll
      for (int n = 0; n < 16; ++n) o[n][r] *= alpha;
      int rowl = r + rowl_base;
      Ps[wave][rowl * 32 + (lane & 15)] = (bf16)p0;
      Ps[wave][rowl * 32 + 16 + (lane & 15)] = (bf16)p1;
    }
#pragma unroll
    for (int ms = 8; ms >= 1; ms >>= 1)
#pragma unroll
      for (int r = 0; r < 8; ++r) tsum[r] += __shfl_xor(tsum[r], ms, 32);
#pragma unroll
    for (int r = 0; r < 8; ++r) lrow[r] += tsum[r];

    // Per-wave LDS round trip for P (C layout -> A layout).
    __builtin_amdgcn_wave_barrier();
    asm volatile("s_wait_dscnt 0" ::: "memory");

    // 5. O += P @ V (K dim = 32 keys, 16 N-tiles over D).
    AFrag pa = load_a16(reinterpret_cast<const unsigned*>(&Ps[wave][0]), 16, 0, 0);
#pragma unroll
    for (int n = 0; n < 16; ++n) {
      AFrag bv = load_b16(reinterpret_cast<const unsigned*>(Vs), 16, n * 16, 0);
      o[n] = wmma_bf16(pa, bv, o[n]);
    }
    __syncthreads();  // all waves done with Ks[buf]/Vs before overwrite
  }

  float inv[8];
#pragma unroll
  for (int r = 0; r < 8; ++r) inv[r] = 1.0f / lrow[r];
  float* Ob = Out + (size_t)b * S * D;
#pragma unroll
  for (int n = 0; n < 16; ++n) {
#pragma unroll
    for (int r = 0; r < 8; ++r) {
      int rowl = r + rowl_base;
      int col = n * 16 + (lane & 15);
      Ob[(size_t)(rbase + rowl) * D + col] = o[n][r] * inv[r];
    }
  }
}

extern "C" void kernel_launch(void* const* d_in, const int* in_sizes, int n_in,
                              void* d_out, int out_size, void* d_ws, size_t ws_size,
                              hipStream_t stream) {
  (void)in_sizes; (void)n_in; (void)out_size; (void)ws_size;
  const float* query = (const float*)d_in[0];
  const float* key   = (const float*)d_in[1];
  const float* value = (const float*)d_in[2];
  const float* Wq    = (const float*)d_in[3];
  const float* Wk    = (const float*)d_in[4];
  const float* Wv    = (const float*)d_in[5];
  float* out = (float*)d_out;

  const size_t N = (size_t)4 * 4096 * 256;
  bf16* qws = (bf16*)d_ws;
  bf16* kws = qws + N;
  bf16* vws = kws + N;

  dim3 gproj(16384 / 128, 3);
  proj_kernel<<<gproj, 256, 0, stream>>>(query, key, value, Wq, Wk, Wv,
                                         qws, kws, vws);

  dim3 gattn(4 * (4096 / 128));
  attn_kernel<<<gattn, 256, 0, stream>>>(qws, kws, vws, out);
}
